// DeformConv2d_58291296141525
// MI455X (gfx1250) — matrix operand order repaired and checked by reading, not compiled, not run on hardware
//
#include <hip/hip_runtime.h>

typedef __attribute__((ext_vector_type(16))) _Float16 v16h;
typedef __attribute__((ext_vector_type(8)))  _Float16 v8h;
typedef __attribute__((ext_vector_type(8)))  float    v8f;

#define BATCH 4
#define CIN   256
#define HH    64
#define WW    64
#define OCH   256
#define PTOT  (HH*WW)     // 4096 pixels per batch image
#define KPTS  9

// ---------------------------------------------------------------------------
// Kernel 1: offset/mask predictor conv (27 output channels, 3x3, pad 1).
// One thread per (b, pixel); 27 accumulators; x value loaded once and reused
// for all 27 output channels. Produces offx, offy, sigmoid(mask) in workspace.
// ---------------------------------------------------------------------------
__global__ __launch_bounds__(256)
void offs_conv(const float* __restrict__ x, const float* __restrict__ ow,
               const float* __restrict__ ob, float* __restrict__ offx,
               float* __restrict__ offy, float* __restrict__ msk) {
    int t = blockIdx.x * blockDim.x + threadIdx.x;   // 0 .. B*P-1
    int p  = t & (PTOT - 1);
    int b  = t >> 12;
    int px = p & (WW - 1);
    int py = p >> 6;

    float acc[27];
#pragma unroll
    for (int ch = 0; ch < 27; ++ch) acc[ch] = ob[ch];

    const float* xb = x + (size_t)b * CIN * PTOT;
    for (int c = 0; c < CIN; ++c) {
        const float* xc = xb + (size_t)c * PTOT;
        for (int kh = 0; kh < 3; ++kh) {
            int yy = py + kh - 1;
            if (yy < 0 || yy >= HH) continue;
            for (int kw = 0; kw < 3; ++kw) {
                int xx = px + kw - 1;
                if (xx < 0 || xx >= WW) continue;
                float xv = xc[yy * WW + xx];
                int wbase = c * 9 + kh * 3 + kw;     // within one out-channel plane
#pragma unroll
                for (int ch = 0; ch < 27; ++ch)
                    acc[ch] = fmaf(xv, ow[ch * (CIN * 9) + wbase], acc[ch]);
            }
        }
    }

    int base = (b * KPTS) * PTOT + p;                // stride PTOT per k
#pragma unroll
    for (int k = 0; k < 9; ++k) {
        offx[base + k * PTOT] = acc[k];              // channels 0..8  -> dx
        offy[base + k * PTOT] = acc[9 + k];          // channels 9..17 -> dy
        float v = acc[18 + k];                       // channels 18..26 -> mask logit
        msk [base + k * PTOT] = 1.0f / (1.0f + __expf(-v));
    }
}

// ---------------------------------------------------------------------------
// Kernel 2: repack fp32 weight [OC, C, 3, 3] into f16 WMMA A-fragments.
// Fragment order: frag = (k*16 + octile)*8 + cchunk; each fragment is
// 32 lanes x 16 halves (1024 B), stored so each lane reads one contiguous
// 32-byte chunk at GEMM time.
// A 16x32 f16 layout (ISA 7.12.2): lane<16: M=lane, K = {0..7, 16..23};
//                                  lane>=16: M=lane-16, K = {8..15, 24..31}.
// ---------------------------------------------------------------------------
__global__ __launch_bounds__(256)
void prepack_w(const float* __restrict__ w, _Float16* __restrict__ apack) {
    int t    = blockIdx.x * blockDim.x + threadIdx.x;  // 0 .. 1152*32-1
    int lane = t & 31;
    int frag = t >> 5;            // 0..1151
    int k      = frag >> 7;       // /128
    int rem    = frag & 127;
    int octile = rem >> 3;        // OC tile (0..15)
    int cchunk = rem & 7;         // 32-wide C chunk (0..7)
    int M     = octile * 16 + (lane & 15);
    int kbase = (lane < 16) ? 0 : 8;

    v16h outv;
#pragma unroll
    for (int e = 0; e < 16; ++e) {
        int Kl = kbase + ((e < 8) ? e : (e + 8));      // K within the 32-chunk
        int c  = cchunk * 32 + Kl;
        outv[e] = (_Float16)w[((size_t)M * CIN + c) * 9 + k];
    }
    *(v16h*)(apack + (size_t)frag * 512 + lane * 16) = outv;
}

// ---------------------------------------------------------------------------
// Kernel 3: fused bilinear sampling + modulated GEMM + bias.
// Grid: (P/16, B). Block: 256 threads = 8 waves.
// WG computes out[b, 0:256, ptile*16 : ptile*16+16].
// Reduction order: for k in 0..8, for cchunk in 0..7 -> WMMA f16 16x16x32.
// Wave w accumulates OC tiles 2w and 2w+1 (f32 accum kept in VGPRs).
// ---------------------------------------------------------------------------
__global__ __launch_bounds__(256)
void dcn_gemm(const float* __restrict__ x, const _Float16* __restrict__ apack,
              const float* __restrict__ offx, const float* __restrict__ offy,
              const float* __restrict__ msk, const float* __restrict__ bias,
              float* __restrict__ out) {
    // S tile: 16 pixels x 256 channels, f16, row stride 264 halves (16B pad
    // per row -> stride 528 B keeps the 16 per-lane b128 reads conflict-free).
    __shared__ __align__(32) _Float16 Sld[16 * 264];
    __shared__ int   sIdx[16][4];
    __shared__ float sWgt[16][4];

    const int ptile = blockIdx.x;     // 0..255
    const int b     = blockIdx.y;
    const int tid   = threadIdx.x;
    const int lane  = tid & 31;
    const int wave  = tid >> 5;       // 0..7
    const int oct0  = wave * 2, oct1 = wave * 2 + 1;
    const int p0    = ptile * 16;

    v8f acc0 = {};
    v8f acc1 = {};

    const float* xb = x + (size_t)b * CIN * PTOT;

    for (int k = 0; k < 9; ++k) {
        __syncthreads();   // previous iteration's WMMA LDS reads complete

        // --- per-pixel bilinear corner indices & modulated weights ---------
        if (tid < 16) {
            int p  = p0 + tid;
            int px = p & 63, py = p >> 6;
            int ob = (b * KPTS + k) * PTOT + p;
            float sx = (float)(px + (k % 3) - 2) + offx[ob]; // px - PAD + (kx-1)
            float sy = (float)(py + (k / 3) - 2) + offy[ob];
            float m  = msk[ob];
            float x0f = floorf(sx), y0f = floorf(sy);
            int   x0  = (int)x0f,   y0  = (int)y0f;
            float wx1 = sx - x0f, wy1 = sy - y0f;
            float wx0 = 1.0f - wx1, wy0 = 1.0f - wy1;
            float cw[4] = {wx0 * wy0, wx1 * wy0, wx0 * wy1, wx1 * wy1};
            int   cx[4] = {x0, x0 + 1, x0, x0 + 1};
            int   cy[4] = {y0, y0, y0 + 1, y0 + 1};
#pragma unroll
            for (int i = 0; i < 4; ++i) {
                bool v = (cx[i] >= 0) && (cx[i] < WW) && (cy[i] >= 0) && (cy[i] < HH);
                sIdx[tid][i] = v ? (cy[i] * WW + cx[i]) : 0;
                sWgt[tid][i] = v ? cw[i] * m : 0.0f;
            }
        }
        __syncthreads();

        // --- cooperative sampling of S_k[256 x 16] into LDS (f16) ----------
        {
            int pix = tid >> 4;        // 0..15
            int cb  = tid & 15;        // channel phase
            int i0 = sIdx[pix][0], i1 = sIdx[pix][1], i2 = sIdx[pix][2], i3 = sIdx[pix][3];
            float w0 = sWgt[pix][0], w1 = sWgt[pix][1], w2 = sWgt[pix][2], w3 = sWgt[pix][3];
#pragma unroll
            for (int i = 0; i < 16; ++i) {
                int c = cb + 16 * i;
                const float* xc = xb + (size_t)c * PTOT;   // L2-resident gathers
                float v = w0 * xc[i0] + w1 * xc[i1] + w2 * xc[i2] + w3 * xc[i3];
                Sld[pix * 264 + c] = (_Float16)v;
            }
        }
        __syncthreads();

        // --- WMMA over 8 chunks of 32 channels -----------------------------
        const _Float16* ak = apack + (size_t)(k * 128) * 512;  // frags for this k
#pragma unroll
        for (int cc = 0; cc < 8; ++cc) {
            int c0 = cc * 32;
            // B 32x16 f16 fragment: lane<16: N=lane, K = {0..7, 16..23};
            //                       lane>=16: N=lane-16, K = {8..15, 24..31}.
            int row = lane & 15;
            int col = c0 + ((lane < 16) ? 0 : 8);
            const _Float16* bp = &Sld[row * 264 + col];
            v8h blo = *(const v8h*)bp;
            v8h bhi = *(const v8h*)(bp + 16);
            v16h bfrag = __builtin_shufflevector(blo, bhi,
                 0, 1, 2, 3, 4, 5, 6, 7, 8, 9, 10, 11, 12, 13, 14, 15);

            // A fragments: one contiguous 32 B per lane, prepacked.
            v16h a0 = *(const v16h*)(ak + (size_t)(oct0 * 8 + cc) * 512 + lane * 16);
            v16h a1 = *(const v16h*)(ak + (size_t)(oct1 * 8 + cc) * 512 + lane * 16);

            acc0 = __builtin_amdgcn_wmma_f32_16x16x32_f16(
                       false, a0, false, bfrag, (short)0, acc0, false, false);
            acc1 = __builtin_amdgcn_wmma_f32_16x16x32_f16(
                       false, a1, false, bfrag, (short)0, acc1, false, false);
        }
    }

    // --- epilogue: bias + store (C/D layout: lane<16 -> M=r, N=lane;
    //               lane>=16 -> M=8+r, N=lane-16) --------------------------
    int n     = lane & 15;
    int mbase = (lane < 16) ? 0 : 8;
    int p     = p0 + n;
#pragma unroll
    for (int r = 0; r < 8; ++r) {
        int oc0 = oct0 * 16 + mbase + r;
        int oc1 = oct1 * 16 + mbase + r;
        out[((size_t)b * OCH + oc0) * PTOT + p] = acc0[r] + bias[oc0];
        out[((size_t)b * OCH + oc1) * PTOT + p] = acc1[r] + bias[oc1];
    }
}

// ---------------------------------------------------------------------------
extern "C" void kernel_launch(void* const* d_in, const int* in_sizes, int n_in,
                              void* d_out, int out_size, void* d_ws, size_t ws_size,
                              hipStream_t stream) {
    const float* x    = (const float*)d_in[0];  // [B,C,H,W]
    const float* ow   = (const float*)d_in[1];  // [27,C,3,3]
    const float* ob   = (const float*)d_in[2];  // [27]
    const float* w    = (const float*)d_in[3];  // [OC,C,3,3]
    const float* bias = (const float*)d_in[4];  // [OC]
    float* out = (float*)d_out;                 // [B,OC,H,W]

    // Workspace layout (≈ 2.95 MB total):
    //   offx / offy / msk : 3 * B*9*P floats  (1.77 MB)
    //   apack             : 1152 frags * 1024 B f16 A-fragments (1.18 MB)
    float* offx = (float*)d_ws;
    float* offy = offx + (size_t)BATCH * KPTS * PTOT;
    float* msk  = offy + (size_t)BATCH * KPTS * PTOT;
    _Float16* apack = (_Float16*)(msk + (size_t)BATCH * KPTS * PTOT);

    offs_conv<<<(BATCH * PTOT) / 256, 256, 0, stream>>>(x, ow, ob, offx, offy, msk);
    prepack_w<<<(9 * 16 * 8 * 32) / 256, 256, 0, stream>>>(w, apack);

    dim3 grid(PTOT / 16, BATCH);
    dcn_gemm<<<grid, 256, 0, stream>>>(x, apack, offx, offy, msk, bias, out);
}
